// MoERouteBlock_25872882991979
// MI455X (gfx1250) — matrix operand order repaired and checked by reading, not compiled, not run on hardware
//
#include <hip/hip_runtime.h>
#include <hip/hip_bf16.h>
#include <math.h>

// ---------------------------------------------------------------- constants
#define BB   16
#define CH   64
#define MIDC 32
#define HH   128
#define WW   128
#define TDIM 128
#define NEXP 4
#define NTOT (BB*CH*HH*WW)          // 16,777,216 output elements

typedef __attribute__((ext_vector_type(16))) _Float16 v16h;
typedef __attribute__((ext_vector_type(8)))  float    v8f;

struct alignas(16) Q16 { unsigned int u[4]; };   // 16-byte copy unit

// Async global->LDS copy of 16 bytes (CDNA5 TDM-class path, ASYNCcnt-tracked).
// ldsOff = wave-relative LDS byte offset (low 32 bits of generic shared ptr:
// flat LDS addressing uses addr[31:0] relative to the wave's LDS base).
__device__ __forceinline__ void async_copy_b128(unsigned ldsOff, const void* gptr)
{
    asm volatile("global_load_async_to_lds_b128 %0, %1, off"
                 :: "v"(ldsOff), "v"(gptr) : "memory");
}
__device__ __forceinline__ void wait_async0()
{
    asm volatile("s_wait_asynccnt 0x0" ::: "memory");
}

// ================================================================ WMMA conv
// Implicit-GEMM 3x3 SAME conv, f16 in / f16 or f32 out, f32 accumulate.
//   in    : f16 NHWC  [B][H][W][CIN]
//   wf    : f16       [COUT][9*CIN]   K-order: (ky*3+kx)*CIN + i, BN-folded
//   bias  : f32       [COUT]          BN-folded
//   res   : f16 NHWC  [B][H][W][COUT] or null (added pre-ReLU)
//   out16 : f16 NHWC  or null
//   out32 : f32 NCHW  or null
// Block: 128 threads (4 waves). Tile: 16 out-channels x 64 pixels of one row.
// grid.x = (W/64)*(COUT/16), grid.y = H, grid.z = B
// CIN is a template parameter so the whole K loop folds to constant LDS
// offsets: straight-line ds_load_b128(imm) + v_wmma, no division in the loop.
template<int CIN>
__global__ __launch_bounds__(128)
void conv3x3_wmma_k(const _Float16* __restrict__ in,
                    const _Float16* __restrict__ wf,
                    const float*    __restrict__ bias,
                    const _Float16* __restrict__ res,
                    _Float16*       __restrict__ out16,
                    float*          __restrict__ out32,
                    int COUT, int relu)
{
    constexpr int CINP   = CIN + 8;      // LDS pad (16B aligned)
    constexpr int KROW   = 9 * CIN;      // GEMM K (576 or 288)
    constexpr int AP     = KROW + 8;     // A row stride in LDS
    constexpr int NCHUNK = KROW / 32;    // 18 or 9 WMMAs per tile

    __shared__ alignas(16) _Float16 sA[16 * AP];
    __shared__ alignas(16) _Float16 sI[3 * 68 * CINP];

    const int tid  = threadIdx.x;
    const int lane = tid & 31;
    const int wave = tid >> 5;

    const int xt = blockIdx.x & 1;              // W/64 == 2
    const int ot = blockIdx.x >> 1;
    const int o0 = ot * 16;
    const int x0 = xt * 64;
    const int y  = blockIdx.y;
    const int b  = blockIdx.z;

    // ---- stage A (weights, 16 x KROW): async global->LDS, no VGPR round-trip
    {
        constexpr int chunks = KROW / 8;
        for (int idx = tid; idx < 16 * chunks; idx += 128) {
            int m = idx / chunks, c = idx - m * chunks;
            async_copy_b128((unsigned)(size_t)&sA[m * AP + c * 8],
                            &wf[(size_t)(o0 + m) * KROW + c * 8]);
        }
    }
    // ---- stage input halo: rows y-1..y+1, x0-1..x0+64, all CIN channels
    {
        constexpr int cch   = CIN / 8;          // 8-half chunks per pixel
        constexpr int total = 3 * 66 * cch;
        for (int idx = tid; idx < total; idx += 128) {
            int c  = idx % cch;
            int t2 = idx / cch;
            int xl = t2 % 66;
            int r  = t2 / 66;
            int yy = y - 1 + r;
            int xg = x0 - 1 + xl;
            const int dst = (r * 68 + xl) * CINP + c * 8;
            if ((unsigned)yy < (unsigned)HH && (unsigned)xg < (unsigned)WW) {
                async_copy_b128((unsigned)(size_t)&sI[dst],
                                &in[(((size_t)b*HH + yy)*WW + xg)*CIN + c*8]);
            } else {
                Q16 z = {{0u,0u,0u,0u}};        // image border: zero-fill
                *(Q16*)&sI[dst] = z;
            }
        }
    }
    wait_async0();          // this wave's async DMAs into LDS are complete
    __syncthreads();        // (compiler adds DScnt wait for the zero stores)

    // ---- K loop: chunks of 32, one v_wmma_f32_16x16x32_f16 each
    v8f acc = {};
    const int nbase = wave * 16;                // this wave's pixel sub-tile
    const int m16   = lane & 15;
    const int hf    = lane >> 4;

    // per-lane fragment bases (everything else is a compile-time immediate)
    const int aBase = m16 * AP + hf * 8;                 // A: 16-bit 16x32 layout
    const int bBase = (nbase + m16) * CINP + hf * 8;     // B: 16-bit 32x16 layout

#pragma unroll
    for (int kk = 0; kk < NCHUNK; ++kk) {
        constexpr int CPT = CIN / 32;           // chunks per tap (2 or 1)
        const int kbase = kk * 32;
        const int p  = kk / CPT;                // tap index (ky*3+kx)
        const int i0 = (kk % CPT) * 32;         // channel base of chunk
        const int ky = p / 3;
        const int kx = p - ky * 3;

        union { v16h v; Q16 q[2]; } A;
        A.q[0] = *(const Q16*)&sA[aBase + kbase];
        A.q[1] = *(const Q16*)&sA[aBase + kbase + 16];

        union { v16h v; Q16 q[2]; } Bf;
        const int boff = (ky * 68 + kx) * CINP + i0;     // constant per kk
        Bf.q[0] = *(const Q16*)&sI[bBase + boff];
        Bf.q[1] = *(const Q16*)&sI[bBase + boff + 16];

        acc = __builtin_amdgcn_wmma_f32_16x16x32_f16(
                  false, A.v, false, Bf.v, (short)0, acc, false, false);
    }

    // ---- epilogue: bias + residual + ReLU, write NHWC f16 and/or NCHW f32
    const int n = lane & 15;
    const int x = x0 + nbase + n;
    const size_t pix = ((size_t)b * HH + y) * WW + x;
#pragma unroll
    for (int j = 0; j < 8; ++j) {
        const int m = j + (hf << 3);            // C/D layout: VGPR j = M j / j+8
        float v = acc[j] + bias[o0 + m];
        if (res)   v += (float)res[pix * COUT + o0 + m];
        if (relu)  v = fmaxf(v, 0.f);
        if (out16) out16[pix * COUT + o0 + m] = (_Float16)v;
        if (out32) out32[(((size_t)b * COUT + o0 + m) * HH + y) * WW + x] = v;
    }
}

// ============================================================ small kernels
// Fold BN into conv weights/bias and repack to [O][(ky*3+kx)*CIN + i] f16.
__global__ void wprep_k(const float* __restrict__ Wc, const float* __restrict__ bc,
                        const float* __restrict__ g,  const float* __restrict__ bt,
                        _Float16* __restrict__ wf, float* __restrict__ bf,
                        int CIN, int COUT)
{
    int idx = blockIdx.x * 256 + threadIdx.x;
    if (idx < COUT)
        bf[idx] = bc[idx] * (g[idx] * rsqrtf(1.f + 1e-5f)) + bt[idx];
    int KROW = 9 * CIN;
    if (idx >= COUT * KROW) return;
    int o = idx / KROW, k = idx - o * KROW;
    int p = k / CIN,    i = k - p * CIN;
    float scale = g[o] * rsqrtf(1.f + 1e-5f);
    wf[idx] = (_Float16)(Wc[(o * CIN + i) * 9 + p] * scale);
}

// feat f32 NCHW -> f16 NHWC, and initialize d_out = feat.
__global__ void convert_feat_k(const float* __restrict__ feat,
                               _Float16* __restrict__ f16,
                               float* __restrict__ dout)
{
    int idx = blockIdx.x * 256 + threadIdx.x;
    if (idx >= NTOT) return;
    int x = idx & 127, y = (idx >> 7) & 127, c = (idx >> 14) & 63, b = idx >> 20;
    float v = feat[idx];
    dout[idx] = v;
    f16[((((size_t)b * HH + y) * WW + x) * CH) + c] = (_Float16)v;
}

// Gating: h=relu(x@g1^T+b), logits=h@g2^T+b, softmax, top-2, R, load, entropy.
__global__ void gate_k(const float* __restrict__ tokens,
                       const float* __restrict__ g1W, const float* __restrict__ g1b,
                       const float* __restrict__ g2W, const float* __restrict__ g2b,
                       float* __restrict__ R, float* __restrict__ tail /* d_out+NTOT */)
{
    __shared__ float h[16][128];
    __shared__ float lg[16][4];
    __shared__ float Rs[16][4];
    int tid = threadIdx.x;
    for (int idx = tid; idx < 16 * 128; idx += 128) {
        int r = idx >> 7, j = idx & 127;
        const float* x = tokens + r * 512;
        const float* w = g1W + j * 512;
        float s = g1b[j];
        for (int k = 0; k < 512; ++k) s = fmaf(x[k], w[k], s);
        h[r][j] = fmaxf(s, 0.f);
    }
    __syncthreads();
    if (tid < 64) {
        int r = tid >> 2, e = tid & 3;
        const float* w = g2W + e * 128;
        float s = g2b[e];
        for (int j = 0; j < 128; ++j) s = fmaf(h[r][j], w[j], s);
        lg[r][e] = s;
    }
    __syncthreads();
    if (tid < 16) {
        int r = tid;
        float mx = lg[r][0];
        for (int e = 1; e < 4; ++e) mx = fmaxf(mx, lg[r][e]);
        float p[4], sum = 0.f;
        for (int e = 0; e < 4; ++e) { p[e] = expf(lg[r][e] - mx); sum += p[e]; }
        for (int e = 0; e < 4; ++e) p[e] /= sum;
        int i1 = 0;
        for (int e = 1; e < 4; ++e) if (p[e] > p[i1]) i1 = e;
        int i2 = (i1 == 0) ? 1 : 0;
        for (int e = 0; e < 4; ++e) if (e != i1 && p[e] > p[i2]) i2 = e;
        for (int e = 0; e < 4; ++e) Rs[r][e] = 0.f;
        Rs[r][i1] = p[i1];
        Rs[r][i2] = p[i2];
        for (int e = 0; e < 4; ++e) R[r * 4 + e] = Rs[r][e];
        ((int*)tail)[5 + r * 2 + 0] = i1;      // topk_idx as raw int32
        ((int*)tail)[5 + r * 2 + 1] = i2;
    }
    __syncthreads();
    if (tid == 0) {
        for (int e = 0; e < 4; ++e) {
            float s = 0.f;
            for (int r = 0; r < 16; ++r) s += Rs[r][e];
            tail[e] = s;                        // load
        }
        float ent = 0.f;
        for (int r = 0; r < 16; ++r) {
            float s = 0.f;
            for (int e = 0; e < 4; ++e) { float v = Rs[r][e]; s += v * logf(v + 1e-9f); }
            ent -= s;
        }
        tail[4] = ent / 16.f;                   // usage_entropy
    }
}

// Y[16][N] = act(X[16][K] @ W[N][K]^T + b)
__global__ void linear16_k(const float* __restrict__ X, const float* __restrict__ Wm,
                           const float* __restrict__ bias, float* __restrict__ Y,
                           int K, int N, int relu)
{
    int idx = blockIdx.x * 128 + threadIdx.x;
    if (idx >= 16 * N) return;
    int r = idx / N, n = idx - r * N;
    const float* x = X + r * K;
    const float* w = Wm + (size_t)n * K;
    float s = bias[n];
    for (int k = 0; k < K; ++k) s = fmaf(x[k], w[k], s);
    if (relu) s = fmaxf(s, 0.f);
    Y[r * N + n] = s;
}

// Y = LayerNorm(A + B) * g + b, rows=16, D=128, one thread per row.
__global__ void add_ln_k(const float* __restrict__ A, const float* __restrict__ Bv,
                         const float* __restrict__ g, const float* __restrict__ b,
                         float* __restrict__ Y)
{
    int r = threadIdx.x;
    if (r >= 16) return;
    const float* a = A + r * 128;
    const float* c = Bv + r * 128;
    float m = 0.f;
    for (int d = 0; d < 128; ++d) m += a[d] + c[d];
    m *= (1.f / 128.f);
    float v = 0.f;
    for (int d = 0; d < 128; ++d) { float t = a[d] + c[d] - m; v += t * t; }
    v *= (1.f / 128.f);
    float inv = rsqrtf(v + 1e-5f);
    for (int d = 0; d < 128; ++d)
        Y[r * 128 + d] = (a[d] + c[d] - m) * inv * g[d] + b[d];
}

// 16x16 average pool: out_e NCHW f32 -> pooled[b][c*64 + gy*8 + gx]
__global__ void pool8_k(const float* __restrict__ oute, float* __restrict__ pooled)
{
    int idx = blockIdx.x * 256 + threadIdx.x;
    if (idx >= BB * CH * 64) return;
    int gidx = idx & 63, c = (idx >> 6) & 63, b = idx >> 12;
    int gy = gidx >> 3, gx = gidx & 7;
    const float* src = oute + (((size_t)b * CH + c) * HH + gy * 16) * WW + gx * 16;
    float s = 0.f;
    for (int yy = 0; yy < 16; ++yy)
        for (int xx = 0; xx < 16; ++xx) s += src[yy * WW + xx];
    pooled[b * 4096 + c * 64 + gidx] = s * (1.f / 256.f);
}

// out_e[b,c,y,x] += bilinear(back[b][c*64..], 8x8 -> 128x128, half-pixel)
__global__ void upsample_add_k(float* __restrict__ oute, const float* __restrict__ back)
{
    int idx = blockIdx.x * 256 + threadIdx.x;
    if (idx >= NTOT) return;
    int x = idx & 127, y = (idx >> 7) & 127, c = (idx >> 14) & 63, b = idx >> 20;
    const float* src = back + b * 4096 + c * 64;
    float sy = (y + 0.5f) * (1.f / 16.f) - 0.5f;
    float sx = (x + 0.5f) * (1.f / 16.f) - 0.5f;
    float fy0 = floorf(sy), fx0 = floorf(sx);
    float wy = sy - fy0, wx = sx - fx0;
    int y0 = max(0, min(7, (int)fy0)), y1 = max(0, min(7, (int)fy0 + 1));
    int x0 = max(0, min(7, (int)fx0)), x1 = max(0, min(7, (int)fx0 + 1));
    float v00 = src[y0 * 8 + x0], v01 = src[y0 * 8 + x1];
    float v10 = src[y1 * 8 + x0], v11 = src[y1 * 8 + x1];
    float v = (1.f - wy) * ((1.f - wx) * v00 + wx * v01)
            +        wy  * ((1.f - wx) * v10 + wx * v11);
    oute[idx] += v;
}

// d_out += out_e * R[b][e]
__global__ void accum_k(float* __restrict__ out, const float* __restrict__ oute,
                        const float* __restrict__ R, int e)
{
    int idx = blockIdx.x * 256 + threadIdx.x;
    if (idx >= NTOT) return;
    int b = idx >> 20;
    out[idx] += oute[idx] * R[b * 4 + e];
}

// ================================================================== launch
extern "C" void kernel_launch(void* const* d_in, const int* in_sizes, int n_in,
                              void* d_out, int out_size, void* d_ws, size_t ws_size,
                              hipStream_t stream)
{
    (void)in_sizes; (void)n_in; (void)out_size; (void)ws_size;
    const float* feat   = (const float*)d_in[0];
    const float* tokens = (const float*)d_in[1];
    // pytree flatten order (dicts sorted by key):
    //  even expert (32 arrays): c1[4] c3[4] ff1_W ff1_b ff2_W ff2_b ftok_W ftok_b
    //    in_W in_b ln1_b ln1_g ln2_b ln2_g out_W out_b r1[4] r2[4] tok_W tok_b
    //  odd expert  (16 arrays): c1[4] c3[4] r1[4] r2[4]
    const int eb[4] = {2, 34, 50, 82};
    const float* g1W = (const float*)d_in[98];
    const float* g1b = (const float*)d_in[99];
    const float* g2W = (const float*)d_in[100];
    const float* g2b = (const float*)d_in[101];

    // ---- workspace carve-up (~137 MB)
    char*  ws  = (char*)d_ws;
    size_t off = 0;
    auto alloc = [&](size_t bytes) -> void* {
        void* p = ws + off;
        off = (off + bytes + 255) & ~(size_t)255;
        return p;
    };
    _Float16* feat16 = (_Float16*)alloc((size_t)NTOT * 2);             // NHWC f16
    _Float16* actA   = (_Float16*)alloc((size_t)BB*HH*WW*MIDC * 2);    // NHWC f16
    _Float16* actB   = (_Float16*)alloc((size_t)BB*HH*WW*MIDC * 2);
    float*    oute   = (float*)   alloc((size_t)NTOT * 4);             // NCHW f32
    _Float16* wf16   = (_Float16*)alloc(64 * 9 * 64 * 2);
    float*    biasf  = (float*)   alloc(64 * 4);
    float*    Rbuf   = (float*)   alloc(BB * NEXP * 4);
    float*    pooled = (float*)   alloc(BB * 4096 * 4);
    float*    tbuf   = (float*)   alloc(BB * 128 * 4);
    float*    vbuf   = (float*)   alloc(BB * 128 * 4);
    float*    abuf   = (float*)   alloc(BB * 128 * 4);
    float*    x1buf  = (float*)   alloc(BB * 128 * 4);
    float*    f1buf  = (float*)   alloc(BB * 256 * 4);
    float*    f2buf  = (float*)   alloc(BB * 128 * 4);
    float*    x2buf  = (float*)   alloc(BB * 128 * 4);
    float*    back   = (float*)   alloc(BB * 4096 * 4);

    float* outp = (float*)d_out;
    float* tail = outp + NTOT;

    const dim3 blk256(256), blk128(128);
    const int  gN = (NTOT + 255) / 256;

    // 1) feat -> f16 NHWC, d_out = feat
    convert_feat_k<<<gN, blk256, 0, stream>>>(feat, feat16, outp);
    // 2) gating (R, load, entropy, topk)
    gate_k<<<1, blk128, 0, stream>>>(tokens, g1W, g1b, g2W, g2b, Rbuf, tail);

    auto prep = [&](int base, int CIN, int COUT) {
        int total = COUT * CIN * 9;
        wprep_k<<<(total + 255) / 256, blk256, 0, stream>>>(
            (const float*)d_in[base], (const float*)d_in[base + 1],
            (const float*)d_in[base + 2], (const float*)d_in[base + 3],
            wf16, biasf, CIN, COUT);
    };
    auto conv64 = [&](const _Float16* in, const _Float16* res,
                      _Float16* o16, float* o32, int COUT, int relu) {
        dim3 g(2 * (COUT / 16), HH, BB);
        conv3x3_wmma_k<64><<<g, blk128, 0, stream>>>(in, wf16, biasf, res,
                                                     o16, o32, COUT, relu);
    };
    auto conv32 = [&](const _Float16* in, const _Float16* res,
                      _Float16* o16, float* o32, int COUT, int relu) {
        dim3 g(2 * (COUT / 16), HH, BB);
        conv3x3_wmma_k<32><<<g, blk128, 0, stream>>>(in, wf16, biasf, res,
                                                     o16, o32, COUT, relu);
    };
    auto lin = [&](const float* X, const float* W, const float* bi,
                   float* Y, int K, int N, int relu) {
        linear16_k<<<(16 * N + 127) / 128, blk128, 0, stream>>>(X, W, bi, Y, K, N, relu);
    };

    for (int e = 0; e < NEXP; ++e) {
        const int base = eb[e];
        const bool tr  = (e % 2 == 0);
        const int r1o  = tr ? 22 : 8;
        const int r2o  = tr ? 26 : 12;

        prep(base + 0, CH, MIDC);                              // c1
        conv64(feat16, nullptr, actA, nullptr, MIDC, 1);
        prep(base + r1o, MIDC, MIDC);                          // r1
        conv32(actA, nullptr, actB, nullptr, MIDC, 1);
        prep(base + r2o, MIDC, MIDC);                          // r2 + residual
        conv32(actB, actA, actA, nullptr, MIDC, 1);
        prep(base + 4, MIDC, CH);                              // c3 (no relu)
        conv32(actA, nullptr, nullptr, oute, CH, 0);

        if (tr) {
            const float* ff1W  = (const float*)d_in[base + 8];
            const float* ff1b  = (const float*)d_in[base + 9];
            const float* ff2W  = (const float*)d_in[base + 10];
            const float* ff2b  = (const float*)d_in[base + 11];
            const float* ftokW = (const float*)d_in[base + 12];
            const float* ftokb = (const float*)d_in[base + 13];
            const float* inW   = (const float*)d_in[base + 14];
            const float* inb   = (const float*)d_in[base + 15];
            const float* ln1b  = (const float*)d_in[base + 16];
            const float* ln1g  = (const float*)d_in[base + 17];
            const float* ln2b  = (const float*)d_in[base + 18];
            const float* ln2g  = (const float*)d_in[base + 19];
            const float* outW  = (const float*)d_in[base + 20];
            const float* outb  = (const float*)d_in[base + 21];
            const float* tokW  = (const float*)d_in[base + 30];
            const float* tokb  = (const float*)d_in[base + 31];

            pool8_k<<<(BB * CH * 64 + 255) / 256, blk256, 0, stream>>>(oute, pooled);
            lin(pooled, tokW, tokb, tbuf, 4096, 128, 0);                 // t
            lin(tbuf, inW + 2 * TDIM * TDIM, inb + 2 * TDIM,
                vbuf, 128, 128, 0);                                      // v slice of qkv
            lin(vbuf, outW, outb, abuf, 128, 128, 0);                    // attn
            add_ln_k<<<1, 16, 0, stream>>>(tbuf, abuf, ln1g, ln1b, x1buf);
            lin(x1buf, ff1W, ff1b, f1buf, 128, 256, 1);
            lin(f1buf, ff2W, ff2b, f2buf, 256, 128, 0);
            add_ln_k<<<1, 16, 0, stream>>>(x1buf, f2buf, ln2g, ln2b, x2buf);
            lin(x2buf, ftokW, ftokb, back, 128, 4096, 0);
            upsample_add_k<<<gN, blk256, 0, stream>>>(oute, back);
        }
        accum_k<<<gN, blk256, 0, stream>>>(outp, oute, Rbuf, e);
    }
}
